// RationalQuadratic_84593675862124
// MI455X (gfx1250) — matrix operand order repaired and checked by reading, not compiled, not run on hardware
//
#include <hip/hip_runtime.h>
#include <hip/hip_bf16.h>
#include <math.h>

// ---------------------------------------------------------------------------
// Fused MLP (bf16 WMMA) + rational-quadratic spline for MI455X (gfx1250).
// - params tensor (411 MB) never touches HBM: produced tile-wise in LDS and
//   consumed immediately by the spline.
// - Weights pre-packed to bf16 in d_ws in this kernel's contiguous-k (k = 16h + e) B-fragment lane layout
//   so each B fragment is two coalesced b128 loads.
// - 32 batch rows per workgroup; each B fragment feeds two WMMAs (2 M-tiles),
//   halving L2->WGP W3 traffic vs 16-row tiles.
// - All 16 A-fragments of H2 are hoisted to registers (128 VGPRs) and reused
//   across all 98 W3 n-tiles: phase-3 inner loop = 2x global_load_b128 + 2 wmma.
// ---------------------------------------------------------------------------

typedef __attribute__((ext_vector_type(16))) __bf16 v16bf;
typedef __attribute__((ext_vector_type(8)))  __bf16 v8bf;
typedef __attribute__((ext_vector_type(8)))  float  v8f;

#define N_BINS   16
#define TAILV    3.0f
#define MIN_VAL  0.001f
#define EPS_V    1e-6f
#define BATCH    65536
#define D1       32
#define DFF      256
#define D2       32
#define PPD      49            // params per dim = 3*16+1
#define DOUT     1568          // PPD * D2
#define HALF_COLS 784          // PPD * 16
#define NT_HALF   49           // 784 / 16
#define ROWS_WG   32           // batch rows per workgroup (2 M-tiles)

// packed-weight offsets inside d_ws (in __bf16 elements)
// tile = 32K x 16N block, 512 bf16: dst[tile*512 + lane*16 + e],
// lane = (kRem/16)*16 + nRem, e = kRem%16  (matches B-frag VGPR layout)
#define PW1_OFF   0                    //  1 kt x 16 nt  =  8192 elems
#define PW2_OFF   8192                 //  8 kt x 16 nt  = 65536 elems
#define PW3_OFF   (8192 + 65536)       //  8 kt x 98 nt  = 401408 elems
// total ws usage: 475136 * 2 B = 950272 bytes

__global__ __launch_bounds__(256)
void rq_pack_weights(const float* __restrict__ src, __bf16* __restrict__ dst,
                     int K, int N) {
    int i = blockIdx.x * blockDim.x + threadIdx.x;
    if (i >= K * N) return;
    int k = i / N, n = i - k * N;
    int kt = k >> 5, kr = k & 31;
    int nt = n >> 4, nr = n & 15;
    int lane = ((kr >> 4) << 4) | nr;
    int e    = kr & 15;
    dst[(size_t)(kt * (N >> 4) + nt) * 512 + lane * 16 + e] = (__bf16)src[i];
}

__device__ __forceinline__ v16bf mk16(v8bf lo, v8bf hi) {
    v16bf r;
#pragma unroll
    for (int e = 0; e < 8; ++e) { r[e] = lo[e]; r[e + 8] = hi[e]; }
    return r;
}

// A fragment: 16x32 bf16 tile from LDS, row-major.
// lane: m = lane&15, g = lane>>4; elems 0..15 = K g*16..g*16+15
__device__ __forceinline__ v16bf load_a(const __bf16* row_base, int g) {
    const __bf16* p = row_base + g * 16;
    v8bf lo = *(const v8bf*)p;
    v8bf hi = *(const v8bf*)(p + 8);
    return mk16(lo, hi);
}

// B fragment: packed tile in global memory, lane's 16 K-values contiguous.
__device__ __forceinline__ v16bf load_b(const __bf16* tile, int lane) {
    const __bf16* p = tile + lane * 16;
    v8bf lo = *(const v8bf*)p;
    v8bf hi = *(const v8bf*)(p + 8);
    return mk16(lo, hi);
}

__device__ __forceinline__ v8f wmma_bf16(v16bf a, v16bf b, v8f c) {
    return __builtin_amdgcn_wmma_f32_16x16x32_bf16(
        false, a, false, b, (short)0, c, false, false);
}

__device__ __forceinline__ float softplus_f(float v) {
    return v > 20.0f ? v : log1pf(__expf(v));
}

// LDS layout (116,864 B total; overlays are phase-disjoint):
//  [0,      16384)   h2s    32x256 bf16
//  [16384,  32768)   h1s    32x256 bf16   (dead before params written)
//  [32768,  34816)   x1s    32x32  bf16   (dead after phase 1)
//  [16384, 116736)   params 32x784 f32    (phase 3/4 only)
//  [116736,116864)   ldsum  32 f32
__global__ __launch_bounds__(256)
void rq_fused_kernel(const float* __restrict__ x1g,
                     const float* __restrict__ x2g,
                     const float* __restrict__ b1g,
                     const float* __restrict__ b2g,
                     const float* __restrict__ b3g,
                     const __bf16* __restrict__ pW1,
                     const __bf16* __restrict__ pW2,
                     const __bf16* __restrict__ pW3,
                     float* __restrict__ z2out,
                     float* __restrict__ ldout) {
    __shared__ __align__(16) unsigned char smem[116864];
    __bf16* h2s    = (__bf16*)(smem);
    __bf16* h1s    = (__bf16*)(smem + 16384);
    __bf16* x1s    = (__bf16*)(smem + 32768);
    float*  params = (float*)(smem + 16384);
    float*  ldsum  = (float*)(smem + 116736);

    const int tid  = threadIdx.x;
    const int lane = tid & 31;
    const int wv   = tid >> 5;
    const int g    = lane >> 4;
    const int nl   = lane & 15;
    const int row0 = blockIdx.x * ROWS_WG;

    // ---- phase 0: stage x1 tile as bf16, zero log-det accumulators --------
#pragma unroll
    for (int i = tid; i < ROWS_WG * D1; i += 256) {
        int r = i >> 5, c = i & 31;
        x1s[i] = (__bf16)x1g[(size_t)(row0 + r) * D1 + c];
    }
    if (tid < ROWS_WG) ldsum[tid] = 0.0f;
    __syncthreads();

    // ---- phase 1: H1 = relu(x1 @ W1 + b1), K = 32 (single k-step) ---------
    for (int task = wv; task < 32; task += 8) {
        const int mt = task >> 4, nt = task & 15;
        v16bf a = load_a(x1s + (mt * 16 + nl) * D1, g);
        v16bf b = load_b(pW1 + (size_t)nt * 512, lane);
        v8f c = {};
        c = wmma_bf16(a, b, c);
        float bias = b1g[nt * 16 + nl];
#pragma unroll
        for (int v = 0; v < 8; ++v) {
            float x = fmaxf(c[v] + bias, 0.0f);
            h1s[(mt * 16 + g * 8 + v) * DFF + nt * 16 + nl] = (__bf16)x;
        }
    }
    __syncthreads();

    // ---- phase 2: H2 = relu(H1 @ W2 + b2), K = 256 (8 k-steps) ------------
    for (int task = wv; task < 32; task += 8) {
        const int mt = task >> 4, nt = task & 15;
        v8f c = {};
#pragma unroll
        for (int kk = 0; kk < 8; ++kk) {
            v16bf a = load_a(h1s + (mt * 16 + nl) * DFF + kk * 32, g);
            v16bf b = load_b(pW2 + (size_t)(kk * 16 + nt) * 512, lane);
            c = wmma_bf16(a, b, c);
        }
        float bias = b2g[nt * 16 + nl];
#pragma unroll
        for (int v = 0; v < 8; ++v) {
            float x = fmaxf(c[v] + bias, 0.0f);
            h2s[(mt * 16 + g * 8 + v) * DFF + nt * 16 + nl] = (__bf16)x;
        }
    }
    __syncthreads();

    // ---- hoist all H2 A-fragments into registers (2 m-tiles x 8 k-steps) --
    v16bf afrag[16];
#pragma unroll
    for (int mt = 0; mt < 2; ++mt)
#pragma unroll
        for (int kk = 0; kk < 8; ++kk)
            afrag[mt * 8 + kk] = load_a(h2s + (mt * 16 + nl) * DFF + kk * 32, g);

    // ---- phases 3+4: per 16-dim half, GEMM3 into LDS then spline ----------
    for (int h = 0; h < 2; ++h) {
        // GEMM3: one B fragment feeds both M-tiles (halves W3 L2 traffic)
        for (int t = wv; t < NT_HALF; t += 8) {
            const int ntg = h * NT_HALF + t;   // global n-tile 0..97
            v8f c0 = {}, c1 = {};
#pragma unroll
            for (int kk = 0; kk < 8; ++kk) {
                const __bf16* tile = pW3 + (size_t)(kk * 98 + ntg) * 512;
                if (kk < 7)
                    __builtin_prefetch(pW3 + (size_t)((kk + 1) * 98 + ntg) * 512, 0, 1);
                v16bf b = load_b(tile, lane);
                c0 = wmma_bf16(afrag[kk],     b, c0);
                c1 = wmma_bf16(afrag[8 + kk], b, c1);
            }
            float bias = b3g[ntg * 16 + nl];
#pragma unroll
            for (int v = 0; v < 8; ++v) {
                params[(g * 8 + v) * HALF_COLS + t * 16 + nl]        = c0[v] + bias;
                params[(16 + g * 8 + v) * HALF_COLS + t * 16 + nl]   = c1[v] + bias;
            }
        }
        __syncthreads();

        // spline: 32 rows x 16 dims = 512 tasks, 2 per thread; reads from LDS
#pragma unroll
        for (int s = 0; s < 2; ++s) {
            const int task = tid + s * 256;
            const int r  = task >> 4;
            const int jj = task & 15;
            const int j  = h * 16 + jj;
            const float* P = params + r * HALF_COLS + jj * PPD;
            const float CNORM = 1.0f - (float)N_BINS * MIN_VAL;

            float mw = -1e30f, mh = -1e30f;
#pragma unroll
            for (int i = 0; i < N_BINS; ++i) {
                mw = fmaxf(mw, P[i]);
                mh = fmaxf(mh, P[N_BINS + i]);
            }
            float sw = 0.0f, sh = 0.0f;
#pragma unroll
            for (int i = 0; i < N_BINS; ++i) {
                sw += __expf(P[i] - mw);
                sh += __expf(P[N_BINS + i] - mh);
            }
            const float iw = CNORM / sw, ih = CNORM / sh;

            const float x = x2g[(size_t)(row0 + r) * D2 + j];
            const bool inside = (x >= -TAILV) && (x <= TAILV);
            const float xc = fminf(fmaxf(x, -TAILV), TAILV);

            // bin search on cumw (ref[last] += EPS, count refs <= xc)
            int idx = 0;
            float cum = 0.0f;
#pragma unroll
            for (int i = 0; i < N_BINS; ++i) {
                cum += MIN_VAL + iw * __expf(P[i] - mw);
                float cw = -TAILV + 2.0f * TAILV * cum;
                if (i == N_BINS - 1) cw += EPS_V;
                if (cw <= xc) idx = i + 1;
            }
            idx = min(idx, N_BINS - 1);

            // capture cumw/cumh bounds of bin idx
            float xk = -TAILV, xk1 = TAILV, yk = -TAILV, yk1 = TAILV;
            float cw_prev = -TAILV, ch_prev = -TAILV;
            float cumw = 0.0f, cumh = 0.0f;
#pragma unroll
            for (int i = 0; i < N_BINS; ++i) {
                cumw += MIN_VAL + iw * __expf(P[i] - mw);
                cumh += MIN_VAL + ih * __expf(P[N_BINS + i] - mh);
                float cw = -TAILV + 2.0f * TAILV * cumw;
                float ch = -TAILV + 2.0f * TAILV * cumh;
                if (i == idx) { xk = cw_prev; xk1 = cw; yk = ch_prev; yk1 = ch; }
                cw_prev = cw; ch_prev = ch;
            }

            const float xkd = xk1 - xk, ykd = yk1 - yk;
            const float sk  = ykd / xkd;
            const float d0  = MIN_VAL + softplus_f(P[2 * N_BINS + idx]);
            const float d1  = MIN_VAL + softplus_f(P[2 * N_BINS + idx + 1]);
            const float xi  = (xc - xk) / xkd;
            const float x1m = xi * (1.0f - xi);
            const float alpha = ykd * (sk * xi * xi + d0 * x1m);
            const float beta  = sk + (d1 + d0 - 2.0f * sk) * x1m;
            const float z     = yk + alpha / beta;
            const float omxi  = 1.0f - xi;
            const float dnum  = sk * sk * (d1 * xi * xi + 2.0f * sk * x1m + d0 * omxi * omxi);
            float ld = __logf(dnum) - 2.0f * __logf(beta);

            z2out[(size_t)(row0 + r) * D2 + j] = inside ? z : x;
            atomicAdd(&ldsum[r], inside ? ld : 0.0f);
        }
        __syncthreads();
    }

    if (tid < ROWS_WG) ldout[row0 + tid] = ldsum[tid];
}

extern "C" void kernel_launch(void* const* d_in, const int* in_sizes, int n_in,
                              void* d_out, int out_size, void* d_ws, size_t ws_size,
                              hipStream_t stream) {
    const float* x1 = (const float*)d_in[0];
    const float* x2 = (const float*)d_in[1];
    const float* W1 = (const float*)d_in[2];
    const float* b1 = (const float*)d_in[3];
    const float* W2 = (const float*)d_in[4];
    const float* b2 = (const float*)d_in[5];
    const float* W3 = (const float*)d_in[6];
    const float* b3 = (const float*)d_in[7];

    __bf16* ws  = (__bf16*)d_ws;
    __bf16* pW1 = ws + PW1_OFF;
    __bf16* pW2 = ws + PW2_OFF;
    __bf16* pW3 = ws + PW3_OFF;

    float* z2out = (float*)d_out;
    float* ldout = (float*)d_out + (size_t)BATCH * D2;

    rq_pack_weights<<<(D1 * DFF + 255) / 256, 256, 0, stream>>>(W1, pW1, D1, DFF);
    rq_pack_weights<<<(DFF * DFF + 255) / 256, 256, 0, stream>>>(W2, pW2, DFF, DFF);
    rq_pack_weights<<<(DFF * DOUT + 255) / 256, 256, 0, stream>>>(W3, pW3, DFF, DOUT);

    rq_fused_kernel<<<BATCH / ROWS_WG, 256, 0, stream>>>(
        x1, x2, b1, b2, b3, pW1, pW2, pW3, z2out, ldout);
}
